// CSRLinear_13597866459289
// MI455X (gfx1250) — matrix operand order repaired and checked by reading, not compiled, not run on hardware
//
#include <hip/hip_runtime.h>
#include <hip/hip_bf16.h>
#include <stdint.h>

// ---------------------------------------------------------------------------
// Sparse linear: out[8192,4096] = x[8192,4096] @ W^T + bias, W given as COO.
// Pipeline:
//   1) zero dense W^T (fp32, K-major) in scratch
//   2) COO scatter-add (global_atomic_add_f32), duplicates sum
//   3) split x -> x_hi/x_lo bf16 planes (row-major [m][k])
//   4) split+transpose W^T -> wb_hi/wb_lo bf16 planes ([n][k], K-major rows)
//   5) bf16x3 WMMA GEMM (hi*hi + hi*lo + lo*hi ~= fp32), 128x128 block tile,
//      double-buffered LDS staging via GLOBAL_LOAD_ASYNC_TO_LDS_B128
//      (ASYNCcnt) when available.
// ---------------------------------------------------------------------------

#define IN_F   4096   // K
#define OUT_F  4096   // N
#define GT_M   128
#define GT_N   128
#define GT_K   32

typedef __attribute__((ext_vector_type(16))) __bf16 v16bf;
typedef __attribute__((ext_vector_type(8)))  float  v8f;
typedef __attribute__((ext_vector_type(4)))  int    v4i;

#if __has_builtin(__builtin_amdgcn_global_load_async_to_lds_b128)
#define ASYNC_LDS 1
#endif

#if defined(ASYNC_LDS)
#if __has_builtin(__builtin_amdgcn_s_wait_asynccnt)
#define WAIT_ASYNC(n) __builtin_amdgcn_s_wait_asynccnt(n)
#else
#define WAIT_ASYNC(n) asm volatile("s_wait_asynccnt %0" ::"i"(n))
#endif
#endif

// Copy 16 bytes global -> LDS (async DMA when available).
__device__ __forceinline__ void cp16(unsigned short* lds, const unsigned short* g) {
#if defined(ASYNC_LDS)
    typedef __attribute__((address_space(1))) v4i* gp_t;   // global int4*
    typedef __attribute__((address_space(3))) v4i* lp_t;   // LDS int4*
    __builtin_amdgcn_global_load_async_to_lds_b128(
        (gp_t)(v4i*)const_cast<unsigned short*>(g),
        (lp_t)(v4i*)lds, 0, 0);
#else
    *reinterpret_cast<uint4*>(lds) = *reinterpret_cast<const uint4*>(g);
#endif
}

// Split fp32 into truncated-bf16 hi and bf16 residual lo (bit patterns).
__device__ __forceinline__ void split_bf16(float v, unsigned& h, unsigned& l) {
    unsigned u  = __float_as_uint(v);
    unsigned hu = u & 0xffff0000u;
    float    hf = __uint_as_float(hu);
    float    rf = v - hf;                 // exact residual
    h = hu >> 16;
    l = __float_as_uint(rf) >> 16;
}

// ---------------- kernel 1: zero the dense W^T scratch ---------------------
__global__ void csrlin_zero_f32(float* __restrict__ p, int n4) {
    int i = blockIdx.x * blockDim.x + threadIdx.x;
    if (i < n4) {
        float4 z; z.x = 0.f; z.y = 0.f; z.z = 0.f; z.w = 0.f;
        reinterpret_cast<float4*>(p)[i] = z;
    }
}

// ---------------- kernel 2: COO scatter-add into W^T (K-major) -------------
__global__ void csrlin_scatter(const float* __restrict__ vals,
                               const int*   __restrict__ rows,
                               const int*   __restrict__ cols,
                               float* __restrict__ Wt, int nnz) {
    int i = blockIdx.x * blockDim.x + threadIdx.x;
    if (i < nnz) {
        atomicAdd(&Wt[(size_t)cols[i] * OUT_F + rows[i]], vals[i]);
    }
}

// ---------------- kernel 3: split x into hi/lo bf16 planes -----------------
__global__ void csrlin_split_x(const float* __restrict__ x,
                               unsigned* __restrict__ xh2,
                               unsigned* __restrict__ xl2, int npairs) {
    int i = blockIdx.x * blockDim.x + threadIdx.x;
    if (i < npairs) {
        float2 v = reinterpret_cast<const float2*>(x)[i];
        unsigned h0, l0, h1, l1;
        split_bf16(v.x, h0, l0);
        split_bf16(v.y, h1, l1);
        xh2[i] = h0 | (h1 << 16);
        xl2[i] = l0 | (l1 << 16);
    }
}

// ------- kernel 4: split + transpose W^T[k][n] -> planes [n][k] ------------
__global__ void csrlin_split_w(const float* __restrict__ Wt,       // [K][N]
                               unsigned short* __restrict__ wh,    // [N][K]
                               unsigned short* __restrict__ wl) {
    __shared__ unsigned short th[32][33];   // +1 pad: no bank conflicts
    __shared__ unsigned short tl[32][33];
    const int kb = blockIdx.x * 32;
    const int nb = blockIdx.y * 32;
    const int t  = threadIdx.x;             // 256
#pragma unroll
    for (int e = 0; e < 4; ++e) {
        int flat = t + e * 256;             // 1024 elems, n fastest (coalesced)
        int n = flat & 31, k = flat >> 5;
        float v = Wt[(size_t)(kb + k) * OUT_F + nb + n];
        unsigned h, l;
        split_bf16(v, h, l);
        th[k][n] = (unsigned short)h;
        tl[k][n] = (unsigned short)l;
    }
    __syncthreads();
#pragma unroll
    for (int e = 0; e < 4; ++e) {
        int flat = t + e * 256;             // k fastest (coalesced out)
        int k = flat & 31, n = flat >> 5;
        wh[(size_t)(nb + n) * IN_F + kb + k] = th[k][n];
        wl[(size_t)(nb + n) * IN_F + kb + k] = tl[k][n];
    }
}

// ---------------- kernel 5: bf16x3 WMMA GEMM -------------------------------
// 256 threads = 8 wave32. Block tile 128(M)x128(N), K-step 32.
// Wave w: rows [16w,16w+16) x all 128 N columns -> 8 f32 accumulators.
__launch_bounds__(256)
__global__ void csrlin_gemm(const unsigned short* __restrict__ xh,  // [M][K]
                            const unsigned short* __restrict__ xl,
                            const unsigned short* __restrict__ wh,  // [N][K]
                            const unsigned short* __restrict__ wl,
                            const float* __restrict__ bias,
                            float* __restrict__ out) {
    // [buf][plane][row][k]; planes: 0=x_hi 1=x_lo 2=w_hi 3=w_lo  (64 KB)
    __shared__ unsigned short smem[2][4][GT_M][GT_K];

    const int t     = threadIdx.x;
    const int lane  = t & 31;
    const int wave  = t >> 5;
    const int nlane = lane & 15;
    const int hi16  = lane >> 4;
    const int mblk  = blockIdx.x * GT_M;
    const int nblk  = blockIdx.y * GT_N;
    const int m0    = wave * 16;

    const unsigned short* gplane[4] = { xh, xl, wh, wl };

    // Accumulators pre-loaded with bias (C/D layout: N = lane&15 both halves).
    v8f acc[8];
#pragma unroll
    for (int j = 0; j < 8; ++j) {
        float b = bias[nblk + j * 16 + nlane];
#pragma unroll
        for (int r = 0; r < 8; ++r) acc[j][r] = b;
    }

    const int akb = hi16 * 8;    // A K-chunk base (ISA 16-bit A layout)
    const int bkb = hi16 * 8;    // B K-chunk base (same layout as A)

    // Stage one 4-plane tile set into LDS buffer `buf` for K offset k0.
    // 4 planes x 512 16B-chunks = 8 chunks per thread.
    auto issue = [&](int buf, int k0) {
#pragma unroll
        for (int p = 0; p < 4; ++p) {
            const unsigned short* base =
                gplane[p] + (size_t)(p < 2 ? mblk : nblk) * IN_F + k0;
#pragma unroll
            for (int c = 0; c < 2; ++c) {
                int chunk = t + c * 256;       // 0..511
                int row   = chunk >> 2;        // 4 chunks per 32-elem row
                int ko    = (chunk & 3) * 8;
                cp16(&smem[buf][p][row][ko], base + (size_t)row * IN_F + ko);
            }
        }
    };

    const int nsteps = IN_F / GT_K;
    issue(0, 0);
    int buf = 0;
    for (int s = 0; s < nsteps; ++s) {
        const bool more = (s + 1 < nsteps);
        if (more) issue(buf ^ 1, (s + 1) * GT_K);   // prefetch next buffer
#if defined(ASYNC_LDS)
        if (more) { WAIT_ASYNC(8); } else { WAIT_ASYNC(0); }  // cur buf done
#endif
        __syncthreads();   // cur buffer visible to all waves

        // A fragments (16x32 bf16): lanes 0-15 M=lane K[0,8)+[16,24);
        // lanes 16-31 M=lane-16 K[8,16)+[24,32) -> two ds_load_b128 each.
        const unsigned short* arow_h = &smem[buf][0][m0 + nlane][0];
        const unsigned short* arow_l = &smem[buf][1][m0 + nlane][0];
        v16bf a_hi, a_lo;
        reinterpret_cast<uint4*>(&a_hi)[0] = *reinterpret_cast<const uint4*>(arow_h + akb);
        reinterpret_cast<uint4*>(&a_hi)[1] = *reinterpret_cast<const uint4*>(arow_h + akb + 16);
        reinterpret_cast<uint4*>(&a_lo)[0] = *reinterpret_cast<const uint4*>(arow_l + akb);
        reinterpret_cast<uint4*>(&a_lo)[1] = *reinterpret_cast<const uint4*>(arow_l + akb + 16);

#pragma unroll
        for (int j = 0; j < 8; ++j) {
            // B fragment (32x16): N=lane&15, K[8h,8h+8)+[16+8h,24+8h), h=lane>=16.
            const unsigned short* bcol_h = &smem[buf][2][j * 16 + nlane][0];
            const unsigned short* bcol_l = &smem[buf][3][j * 16 + nlane][0];
            v16bf b_hi, b_lo;
            reinterpret_cast<uint4*>(&b_hi)[0] = *reinterpret_cast<const uint4*>(bcol_h + bkb);
            reinterpret_cast<uint4*>(&b_hi)[1] = *reinterpret_cast<const uint4*>(bcol_h + bkb + 16);
            reinterpret_cast<uint4*>(&b_lo)[0] = *reinterpret_cast<const uint4*>(bcol_l + bkb);
            reinterpret_cast<uint4*>(&b_lo)[1] = *reinterpret_cast<const uint4*>(bcol_l + bkb + 16);

            acc[j] = __builtin_amdgcn_wmma_f32_16x16x32_bf16(
                false, a_hi, false, b_hi, (short)0, acc[j], false, false);
            acc[j] = __builtin_amdgcn_wmma_f32_16x16x32_bf16(
                false, a_hi, false, b_lo, (short)0, acc[j], false, false);
            acc[j] = __builtin_amdgcn_wmma_f32_16x16x32_bf16(
                false, a_lo, false, b_hi, (short)0, acc[j], false, false);
        }
        __syncthreads();   // all waves done reading -> safe to overwrite
        buf ^= 1;
    }

    // Writeback: element (M = r + 8*hi16, N = lane&15) per C/D layout.
#pragma unroll
    for (int j = 0; j < 8; ++j) {
#pragma unroll
        for (int r = 0; r < 8; ++r) {
            out[(size_t)(mblk + m0 + hi16 * 8 + r) * OUT_F + nblk + j * 16 + nlane]
                = acc[j][r];
        }
    }
}

// ---------------------------------------------------------------------------
extern "C" void kernel_launch(void* const* d_in, const int* in_sizes, int n_in,
                              void* d_out, int out_size, void* d_ws, size_t ws_size,
                              hipStream_t stream) {
    const float* x    = (const float*)d_in[0];
    const float* vals = (const float*)d_in[1];
    const int*   rows = (const int*)d_in[2];
    const int*   cols = (const int*)d_in[3];
    const float* bias = (const float*)d_in[4];
    float*       outp = (float*)d_out;

    const int nnz    = in_sizes[1];
    const int tokens = in_sizes[0] / IN_F;   // 8192

    // Scratch layout (256 MB total):
    //   [0,64M)    Wt   fp32 [K][N]
    //   [64,128M)  x_hi bf16 [M][K]
    //   [128,192M) x_lo bf16 [M][K]
    //   [192,224M) wb_hi bf16 [N][K]
    //   [224,256M) wb_lo bf16 [N][K]
    char* ws = (char*)d_ws;
    float*          Wt  = (float*)ws;
    unsigned short* xh  = (unsigned short*)(ws + (size_t)64 * 1024 * 1024);
    unsigned short* xlo = (unsigned short*)(ws + (size_t)128 * 1024 * 1024);
    unsigned short* wbh = (unsigned short*)(ws + (size_t)192 * 1024 * 1024);
    unsigned short* wbl = (unsigned short*)(ws + (size_t)224 * 1024 * 1024);

    // 1) zero W^T
    {
        int n4 = (IN_F * OUT_F) / 4;
        csrlin_zero_f32<<<(n4 + 255) / 256, 256, 0, stream>>>(Wt, n4);
    }
    // 2) COO scatter-add
    csrlin_scatter<<<(nnz + 255) / 256, 256, 0, stream>>>(vals, rows, cols, Wt, nnz);
    // 3) split x
    {
        int npairs = tokens * IN_F / 2;
        csrlin_split_x<<<(npairs + 255) / 256, 256, 0, stream>>>(
            x, (unsigned*)xh, (unsigned*)xlo, npairs);
    }
    // 4) split + transpose W
    {
        dim3 g(IN_F / 32, OUT_F / 32);
        csrlin_split_w<<<g, 256, 0, stream>>>(Wt, wbh, wbl);
    }
    // 5) bf16x3 WMMA GEMM
    {
        dim3 grid(tokens / GT_M, OUT_F / GT_N);
        csrlin_gemm<<<grid, 256, 0, stream>>>(xh, xlo, wbh, wbl, bias, outp);
    }
}
